// MultiheadAttention_2353642078626
// MI455X (gfx1250) — matrix operand order repaired and checked by reading, not compiled, not run on hardware
//
#include <hip/hip_runtime.h>
#include <hip/hip_bf16.h>

#define B_  4
#define S_  2048
#define D_  1024
#define H_  16
#define HD_ 64

typedef __bf16 bf16_t;
typedef __attribute__((ext_vector_type(16))) __bf16 v16bf;
typedef __attribute__((ext_vector_type(8)))  __bf16 v8bf;
typedef __attribute__((ext_vector_type(4)))  __bf16 v4bf;
typedef __attribute__((ext_vector_type(8)))  float  v8f;
typedef __attribute__((ext_vector_type(4)))  unsigned int u32x4;
typedef __attribute__((ext_vector_type(8)))  int  i32x8;
typedef __attribute__((ext_vector_type(4)))  int  i32x4;

// ---------------------------------------------------------------------------
// Tensor Data Mover: 2D tile load Global -> LDS (D# per CDNA5 ISA 8.3/8.4).
// ---------------------------------------------------------------------------
__device__ __forceinline__ void tdm_load_2d(unsigned lds_off, const void* gaddr,
                                            unsigned tensor_d0, unsigned tensor_d1,
                                            unsigned tile_d0, unsigned tile_d1,
                                            unsigned long long stride0_elems) {
    unsigned long long ga = (unsigned long long)gaddr;
    u32x4 g0;
    g0[0] = 1u;                                               // count=1, user descriptor
    g0[1] = lds_off;                                          // lds_addr (bytes)
    g0[2] = (unsigned)(ga & 0xFFFFFFFFu);                     // global_addr[31:0]
    g0[3] = (unsigned)((ga >> 32) & 0x1FFFFFFu) | (2u << 30); // global_addr[56:32] | type=2
    i32x8 g1;
    g1[0] = (int)(1u << 16);                                  // data_size=1 (2B elements)
    g1[1] = (int)((tensor_d0 & 0xFFFFu) << 16);               // tensor_dim0[15:0]
    g1[2] = (int)((tensor_d0 >> 16) | ((tensor_d1 & 0xFFFFu) << 16));
    g1[3] = (int)((tensor_d1 >> 16) | (tile_d0 << 16));       // tile_dim0
    g1[4] = (int)(tile_d1);                                   // tile_dim1 (tile_dim2=0)
    g1[5] = (int)(stride0_elems & 0xFFFFFFFFull);             // dim0 stride lo32
    g1[6] = (int)((stride0_elems >> 32) & 0xFFFFull);         // dim0 stride hi16
    g1[7] = 0;
    i32x4 z4 = {0, 0, 0, 0};
#if defined(__clang_major__) && (__clang_major__ >= 23)
    i32x8 z8 = {0, 0, 0, 0, 0, 0, 0, 0};
    __builtin_amdgcn_tensor_load_to_lds(g0, g1, z4, z4, z8, 0);
#else
    __builtin_amdgcn_tensor_load_to_lds(g0, g1, z4, z4, 0);
#endif
}

// ---------------------------------------------------------------------------
// f32 -> bf16 conversion (4 elements / thread)
// ---------------------------------------------------------------------------
__global__ void cvt_f32_bf16(const float* __restrict__ in, bf16_t* __restrict__ out, int n) {
    int i = (blockIdx.x * blockDim.x + threadIdx.x) * 4;
    if (i < n) {
        float4 f = *reinterpret_cast<const float4*>(in + i);
        v4bf r;
        r[0] = (bf16_t)f.x; r[1] = (bf16_t)f.y; r[2] = (bf16_t)f.z; r[3] = (bf16_t)f.w;
        *reinterpret_cast<v4bf*>(out + i) = r;
    }
}

// ---------------------------------------------------------------------------
// V transpose: Vb [B,S,D] (head-interleaved) -> VT [B,H,HD,S]
// ---------------------------------------------------------------------------
__global__ void transpose_v(const bf16_t* __restrict__ v, bf16_t* __restrict__ vt) {
    int idx = blockIdx.x * blockDim.x + threadIdx.x;   // over B*H*HD*S = 2^23
    int s = idx & (S_ - 1);
    int d = (idx >> 11) & (HD_ - 1);
    int h = (idx >> 17) & (H_ - 1);
    int b = idx >> 21;
    vt[idx] = v[((size_t)(b * S_ + s)) * D_ + h * HD_ + d];
}

// ---------------------------------------------------------------------------
// GEMM: out[M,N] = A[M,K](bf16) @ W[N,K]^T(bf16) + bias[N](f32)
// Block: 256 threads = 8 waves (2 M x 4 N), wave tile 16x64, block tile 32x256.
// K pipelined in 64-deep steps with double-buffered TDM tiles:
//   A tile 32x64 (4 KB), B tile 256x64 (32 KB), x2 buffers = 72 KB LDS.
// Wave0/wave1 issue the next step's TENSOR_LOAD_TO_LDS, then s_wait_tensorcnt(1)
// (TDM ops of a wave complete in order -> the older load is done, the new one
// overlaps the 8x8 WMMA compute phase).
// ---------------------------------------------------------------------------
template <bool OUTF32>
__global__ __launch_bounds__(256, 2)
void gemm_wmma(const bf16_t* __restrict__ A, const bf16_t* __restrict__ W,
               const float* __restrict__ bias, void* __restrict__ outv,
               int M, int N, int K) {
    __shared__ __align__(32) bf16_t Abuf[2][32][64];      //  8 KB
    __shared__ __align__(32) bf16_t Bbuf[2][256][64];     // 64 KB

    const int lane = threadIdx.x & 31;
    const int w    = threadIdx.x >> 5;
    const int wm   = w >> 2;
    const int wn   = w & 3;
    const int mblk = blockIdx.y * 32;
    const int nblk = blockIdx.x * 256;
    const int m0   = mblk + wm * 16;
    const int n0   = nblk + wn * 64;
    const int l15  = lane & 15;
    const int hi   = (lane >> 4) & 1;
    const int a0   = hi * 16;
    const int b0   = hi * 16;

    v8f acc[4] = {};

    const int nsteps = K >> 6;        // K/64

    // prologue: stage first K-tile into buffer 0
    if (w == 0) {
        tdm_load_2d((unsigned)(size_t)&Abuf[0][0][0], A + (size_t)mblk * K,
                    (unsigned)K, 32u, 64u, 32u, (unsigned long long)K);
    } else if (w == 1) {
        tdm_load_2d((unsigned)(size_t)&Bbuf[0][0][0], W + (size_t)nblk * K,
                    (unsigned)K, 256u, 64u, 256u, (unsigned long long)K);
    }

    for (int i = 0; i < nsteps; ++i) {
        const int cb = i & 1;
        // issue next tile into the other buffer, then wait for the current one
        if (w == 0) {
            if (i + 1 < nsteps) {
                tdm_load_2d((unsigned)(size_t)&Abuf[cb ^ 1][0][0],
                            A + (size_t)mblk * K + (i + 1) * 64,
                            (unsigned)(K - (i + 1) * 64), 32u, 64u, 32u,
                            (unsigned long long)K);
                __builtin_amdgcn_s_wait_tensorcnt(1);
            } else {
                __builtin_amdgcn_s_wait_tensorcnt(0);
            }
        } else if (w == 1) {
            if (i + 1 < nsteps) {
                tdm_load_2d((unsigned)(size_t)&Bbuf[cb ^ 1][0][0],
                            W + (size_t)nblk * K + (i + 1) * 64,
                            (unsigned)(K - (i + 1) * 64), 256u, 64u, 256u,
                            (unsigned long long)K);
                __builtin_amdgcn_s_wait_tensorcnt(1);
            } else {
                __builtin_amdgcn_s_wait_tensorcnt(0);
            }
        }
        __syncthreads();   // current tiles visible to all 8 waves

#pragma unroll
        for (int s = 0; s < 2; ++s) {          // two 32-deep WMMA k-steps
            const bf16_t* ar = &Abuf[cb][wm * 16 + l15][s * 32];
            v8bf alo = *reinterpret_cast<const v8bf*>(ar + a0);
            v8bf ahi = *reinterpret_cast<const v8bf*>(ar + a0 + 8);
            v16bf a;
#pragma unroll
            for (int j = 0; j < 8; ++j) { a[j] = alo[j]; a[j + 8] = ahi[j]; }
#pragma unroll
            for (int t = 0; t < 4; ++t) {
                const bf16_t* br = &Bbuf[cb][wn * 64 + t * 16 + l15][s * 32];
                v16bf bvec = *reinterpret_cast<const v16bf*>(br + b0);
                acc[t] = __builtin_amdgcn_wmma_f32_16x16x32_bf16(
                    false, a, false, bvec, (short)0, acc[t], false, false);
            }
        }
        __syncthreads();   // all reads done before this buffer is overwritten
    }

#pragma unroll
    for (int t = 0; t < 4; ++t) {
        int n = n0 + t * 16 + l15;
        float bv = bias[n];
        if (OUTF32) {
            float* out = (float*)outv;
#pragma unroll
            for (int e = 0; e < 8; ++e)
                out[(size_t)(m0 + e + hi * 8) * N + n] = acc[t][e] + bv;
        } else {
            bf16_t* out = (bf16_t*)outv;
#pragma unroll
            for (int e = 0; e < 8; ++e)
                out[(size_t)(m0 + e + hi * 8) * N + n] = (bf16_t)(acc[t][e] + bv);
        }
    }
}

// ---------------------------------------------------------------------------
// Flash attention (causal), one wave per 16-query tile, key blocks of 32.
// Causal mask applied only on the diagonal key block; row sums computed with
// one WMMA against an all-ones B fragment instead of lane shuffles.
// Q,K: [B,S,D] bf16 (head-interleaved). VT: [B,H,HD,S] bf16. C: [B,S,D] bf16.
// ---------------------------------------------------------------------------
__global__ __launch_bounds__(256, 2)
void flash_attn(const bf16_t* __restrict__ Q, const bf16_t* __restrict__ Km,
                const bf16_t* __restrict__ VT, bf16_t* __restrict__ C) {
    __shared__ __align__(32) bf16_t plds[8][16 * 32];

    const int lane = threadIdx.x & 31;
    const int w    = threadIdx.x >> 5;
    const int gw   = blockIdx.x * 8 + w;        // global wave id
    const int qt   = gw & 127;                  // S/16 = 128 query tiles
    const int h    = (gw >> 7) & 15;
    const int b    = gw >> 11;
    const int q0   = qt * 16;
    const int l15  = lane & 15;
    const int hi   = (lane >> 4) & 1;
    const int a0   = hi * 16;                   // A-frag K sub-offset
    const int b0   = hi * 16;                   // B-frag K sub-offset

    // --- load the two Q A-fragments (K-steps d=0..31 and d=32..63), kept live
    const bf16_t* qp = Q + (size_t)(b * S_ + q0 + l15) * D_ + h * HD_;
    v16bf qa0, qa1;
    {
        v8bf lo = *reinterpret_cast<const v8bf*>(qp + a0);
        v8bf hv = *reinterpret_cast<const v8bf*>(qp + a0 + 8);
#pragma unroll
        for (int i = 0; i < 8; ++i) { qa0[i] = lo[i]; qa0[i + 8] = hv[i]; }
        lo = *reinterpret_cast<const v8bf*>(qp + 32 + a0);
        hv = *reinterpret_cast<const v8bf*>(qp + 32 + a0 + 8);
#pragma unroll
        for (int i = 0; i < 8; ++i) { qa1[i] = lo[i]; qa1[i + 8] = hv[i]; }
    }

    // constant all-ones B fragment for row-sum WMMA
    v16bf ones;
#pragma unroll
    for (int i = 0; i < 16; ++i) ones[i] = (bf16_t)1.0f;

    v8f acc0 = {}, acc1 = {}, acc2 = {}, acc3 = {};
    float mrow[8], lrow[8], alpha8[8];
#pragma unroll
    for (int e = 0; e < 8; ++e) { mrow[e] = -3.0e38f; lrow[e] = 0.f; }

    const int nkb = q0 / 32 + 1;                // causal key-block count
    for (int kb = 0; kb < nkb; ++kb) {
        const int ks = kb * 32;

        // speculative prefetch of next key block
        if (kb + 1 < nkb) {
            const bf16_t* kpn = Km + (size_t)(b * S_ + ks + 32 + l15) * D_ + h * HD_;
            __builtin_prefetch((const void*)kpn, 0, 1);
        }

        // ---- scores: two 16-column tiles, each 2 WMMA k-steps over HD=64
        v8f sc[2];
#pragma unroll
        for (int nt = 0; nt < 2; ++nt) {
            const bf16_t* kp = Km + (size_t)(b * S_ + ks + nt * 16 + l15) * D_ + h * HD_;
            v16bf kf0 = *reinterpret_cast<const v16bf*>(kp + b0);
            v16bf kf1 = *reinterpret_cast<const v16bf*>(kp + 32 + b0);
            v8f s = {};
            s = __builtin_amdgcn_wmma_f32_16x16x32_bf16(false, qa0, false, kf0,
                                                        (short)0, s, false, false);
            s = __builtin_amdgcn_wmma_f32_16x16x32_bf16(false, qa1, false, kf1,
                                                        (short)0, s, false, false);
            sc[nt] = s;
        }

        // ---- scale; causal mask needed only on the diagonal block
        if (kb == nkb - 1) {
#pragma unroll
            for (int e = 0; e < 8; ++e) {
                const int row = q0 + e + hi * 8;
#pragma unroll
                for (int nt = 0; nt < 2; ++nt) {
                    const int col = ks + nt * 16 + l15;
                    float v = sc[nt][e] * 0.125f;
                    sc[nt][e] = (col <= row) ? v : -3.0e38f;
                }
            }
        } else {
#pragma unroll
            for (int e = 0; e < 8; ++e) {
                sc[0][e] *= 0.125f;
                sc[1][e] *= 0.125f;
            }
        }

        // ---- online softmax: row max over the 16-lane half, then exp
#pragma unroll
        for (int e = 0; e < 8; ++e) {
            float r = fmaxf(sc[0][e], sc[1][e]);
#pragma unroll
            for (int mm = 1; mm < 16; mm <<= 1) r = fmaxf(r, __shfl_xor(r, mm, 32));
            const float mnew = fmaxf(mrow[e], r);
            alpha8[e] = __expf(mrow[e] - mnew);
            sc[0][e] = __expf(sc[0][e] - mnew);
            sc[1][e] = __expf(sc[1][e] - mnew);
            mrow[e] = mnew;
            acc0[e] *= alpha8[e]; acc1[e] *= alpha8[e];
            acc2[e] *= alpha8[e]; acc3[e] *= alpha8[e];
        }

        // ---- C-layout -> A-layout transpose of P through per-wave LDS tile
#pragma unroll
        for (int nt = 0; nt < 2; ++nt)
#pragma unroll
            for (int e = 0; e < 8; ++e)
                plds[w][(e + hi * 8) * 32 + nt * 16 + l15] = (bf16_t)sc[nt][e];
        __builtin_amdgcn_wave_barrier();
        v16bf pa;
        {
            const bf16_t* pp = &plds[w][l15 * 32 + a0];
            v8bf lo = *reinterpret_cast<const v8bf*>(pp);
            v8bf hv = *reinterpret_cast<const v8bf*>(pp + 8);
#pragma unroll
            for (int i = 0; i < 8; ++i) { pa[i] = lo[i]; pa[i + 8] = hv[i]; }
        }
        __builtin_amdgcn_wave_barrier();

        // ---- row sums via WMMA: P(16x32) @ ones(32x16) -> every lane holds
        //      the sum of its 8 rows in C layout (matches lrow indexing)
        {
            v8f lsum = {};
            lsum = __builtin_amdgcn_wmma_f32_16x16x32_bf16(false, pa, false, ones,
                                                           (short)0, lsum, false, false);
#pragma unroll
            for (int e = 0; e < 8; ++e)
                lrow[e] = lrow[e] * alpha8[e] + lsum[e];
        }

        // ---- ctx += P(16x32) @ Vblk(32x64): 4 column tiles from VT
        const bf16_t* vp = VT + ((size_t)((b * H_ + h) * HD_) + l15) * S_ + ks + b0;
        {
            v16bf v0 = *reinterpret_cast<const v16bf*>(vp);
            v16bf v1 = *reinterpret_cast<const v16bf*>(vp + (size_t)16 * S_);
            v16bf v2 = *reinterpret_cast<const v16bf*>(vp + (size_t)32 * S_);
            v16bf v3 = *reinterpret_cast<const v16bf*>(vp + (size_t)48 * S_);
            acc0 = __builtin_amdgcn_wmma_f32_16x16x32_bf16(false, pa, false, v0,
                                                           (short)0, acc0, false, false);
            acc1 = __builtin_amdgcn_wmma_f32_16x16x32_bf16(false, pa, false, v1,
                                                           (short)0, acc1, false, false);
            acc2 = __builtin_amdgcn_wmma_f32_16x16x32_bf16(false, pa, false, v2,
                                                           (short)0, acc2, false, false);
            acc3 = __builtin_amdgcn_wmma_f32_16x16x32_bf16(false, pa, false, v3,
                                                           (short)0, acc3, false, false);
        }
    }

    // ---- normalize and store ctx tile (bf16, [B,S,D] head-interleaved)
    bf16_t* cp = C + (size_t)(b * S_ + q0) * D_ + h * HD_;
#pragma unroll
    for (int e = 0; e < 8; ++e) {
        const float inv = 1.0f / lrow[e];
        bf16_t* cr = cp + (size_t)(e + hi * 8) * D_;
        cr[l15]      = (bf16_t)(acc0[e] * inv);
        cr[16 + l15] = (bf16_t)(acc1[e] * inv);
        cr[32 + l15] = (bf16_t)(acc2[e] * inv);
        cr[48 + l15] = (bf16_t)(acc3[e] * inv);
    }
}

// ---------------------------------------------------------------------------
// Host-side launcher
// ---------------------------------------------------------------------------
extern "C" void kernel_launch(void* const* d_in, const int* in_sizes, int n_in,
                              void* d_out, int out_size, void* d_ws, size_t ws_size,
                              hipStream_t stream) {
    (void)in_sizes; (void)n_in; (void)out_size; (void)ws_size;
    const float* x  = (const float*)d_in[0];
    const float* Wq = (const float*)d_in[1];
    const float* bq = (const float*)d_in[2];
    const float* Wk = (const float*)d_in[3];
    const float* bk = (const float*)d_in[4];
    const float* Wv = (const float*)d_in[5];
    const float* bv = (const float*)d_in[6];
    const float* Wo = (const float*)d_in[7];
    const float* bo = (const float*)d_in[8];

    char* ws = (char*)d_ws;
    const size_t MB = 1ull << 20;
    bf16_t* xb  = (bf16_t*)(ws + 0);        // 16 MB
    bf16_t* Wqb = (bf16_t*)(ws + 16 * MB);  // 2 MB
    bf16_t* Wkb = (bf16_t*)(ws + 18 * MB);
    bf16_t* Wvb = (bf16_t*)(ws + 20 * MB);
    bf16_t* Wob = (bf16_t*)(ws + 22 * MB);
    bf16_t* Qb  = (bf16_t*)(ws + 24 * MB);  // 16 MB
    bf16_t* Kb  = (bf16_t*)(ws + 40 * MB);  // 16 MB
    bf16_t* Vb  = (bf16_t*)(ws + 56 * MB);  // 16 MB
    bf16_t* VTb = (bf16_t*)(ws + 72 * MB);  // 16 MB
    bf16_t* Cb  = (bf16_t*)(ws + 88 * MB);  // 16 MB  (total 104 MB)

    const int NX = B_ * S_ * D_;   // 8388608
    const int NW = D_ * D_;        // 1048576

    cvt_f32_bf16<<<NX / 1024, 256, 0, stream>>>(x,  xb,  NX);
    cvt_f32_bf16<<<NW / 1024, 256, 0, stream>>>(Wq, Wqb, NW);
    cvt_f32_bf16<<<NW / 1024, 256, 0, stream>>>(Wk, Wkb, NW);
    cvt_f32_bf16<<<NW / 1024, 256, 0, stream>>>(Wv, Wvb, NW);
    cvt_f32_bf16<<<NW / 1024, 256, 0, stream>>>(Wo, Wob, NW);

    dim3 gg(D_ / 256, (B_ * S_) / 32);     // (4, 256)
    gemm_wmma<false><<<gg, 256, 0, stream>>>(xb, Wqb, bq, (void*)Qb, B_ * S_, D_, D_);
    gemm_wmma<false><<<gg, 256, 0, stream>>>(xb, Wkb, bk, (void*)Kb, B_ * S_, D_, D_);
    gemm_wmma<false><<<gg, 256, 0, stream>>>(xb, Wvb, bv, (void*)Vb, B_ * S_, D_, D_);

    transpose_v<<<NX / 256, 256, 0, stream>>>(Vb, VTb);

    flash_attn<<<(B_ * H_ * (S_ / 16)) / 8, 256, 0, stream>>>(Qb, Kb, VTb, Cb);

    gemm_wmma<true><<<gg, 256, 0, stream>>>(Cb, Wob, bo, d_out, B_ * S_, D_, D_);
}
